// ContrastLoss3_26731876450777
// MI455X (gfx1250) — matrix operand order repaired and checked by reading, not compiled, not run on hardware
//
#include <hip/hip_runtime.h>
#include <hip/hip_bf16.h>

// Problem constants (from reference)
#define B_SMP   256
#define T_SLOT  27
#define N_TOK   6912          // B_SMP * T_SLOT
#define DIM     1024
#define NCS     27            // column chunks (6912 / 256)
#define COLS_PER_CHUNK 256
#define ROWS_PER_BLOCK 256
#define ROWS_PER_WAVE  64

// ws layout (bytes):
//   g       : N_TOK*DIM f16            @ 0            (14,155,776)
#define WS_OFF_COLLAB  14155776u        // N_TOK int        (27,648)
#define WS_OFF_POS     14183424u        // N_TOK*NCS f32   (746,496)
#define WS_OFF_NEG     14929920u        // N_TOK*NCS f32   (746,496)

typedef __attribute__((ext_vector_type(16))) _Float16 v16h;
typedef __attribute__((ext_vector_type(8)))  _Float16 v8h;
typedef __attribute__((ext_vector_type(4)))  _Float16 v4h;
typedef __attribute__((ext_vector_type(8)))  float    v8f;

// ---------------------------------------------------------------------------
// P1: ragged prefix mask -> per-row label array (label if valid else -1)
// fz = argmax(target[:,1:]==0) + 1  (first zero; 1 if none). valid: t < fz.
// ---------------------------------------------------------------------------
__global__ void prep_labels(const int* __restrict__ tgt, int* __restrict__ colLab) {
    int b = blockIdx.x * blockDim.x + threadIdx.x;
    if (b >= B_SMP) return;
    const int* row = tgt + b * T_SLOT;
    int fz = 1;
    for (int t = 1; t < T_SLOT; ++t) {
        if (row[t] == 0) { fz = t; break; }
    }
    for (int t = 0; t < T_SLOT; ++t)
        colLab[b * T_SLOT + t] = (t < fz) ? row[t] : -1;
}

// ---------------------------------------------------------------------------
// P2: L2-normalize each row (fp32 math), emit f16 row-major matrix g[N][D].
// One wave32 per row; 8 waves per 256-thread block.
// ---------------------------------------------------------------------------
__global__ void prep_norm(const float* __restrict__ f, _Float16* __restrict__ g) {
    int row  = (int)((blockIdx.x * blockDim.x + threadIdx.x) >> 5);
    int lane = threadIdx.x & 31;
    if (row >= N_TOK) return;

    const float4* fr = (const float4*)(f + (size_t)row * DIM);
    float4 v[8];
    float ss = 0.f;
#pragma unroll
    for (int j = 0; j < 8; ++j) {
        v[j] = fr[j * 32 + lane];
        ss += v[j].x * v[j].x + v[j].y * v[j].y + v[j].z * v[j].z + v[j].w * v[j].w;
    }
#pragma unroll
    for (int off = 16; off >= 1; off >>= 1)
        ss += __shfl_xor(ss, off, 32);

    float sc = 1.0f / fmaxf(sqrtf(ss), 1e-12f);

    v4h* gr = (v4h*)(g + (size_t)row * DIM);
#pragma unroll
    for (int j = 0; j < 8; ++j) {
        v4h h;
        h[0] = (_Float16)(v[j].x * sc);
        h[1] = (_Float16)(v[j].y * sc);
        h[2] = (_Float16)(v[j].z * sc);
        h[3] = (_Float16)(v[j].w * sc);
        gr[j * 32 + lane] = h;
    }
}

// ---------------------------------------------------------------------------
// Main fused GEMM + masked exp reduction.
// Block = 4 waves (128 threads). Wave tile = 64 rows x 64 cols (4x4 WMMA
// fragments, f16 in / f32 acc). grid = (27 row blocks, 27 col chunks)
// -> 2916 waves, ~3 waves/SIMD, matching the ~320-VGPR occupancy ceiling.
// A fragment (16x32 f16): lane<16 -> M=lane, K = kk+0..15 (contiguous)
//                         lane>=16 -> M=lane-16, K = kk+16..31
// B fragment (32x16 f16): lane<16 -> N=lane, K = kk+0..15 (contiguous)
//                         lane>=16 -> N=lane-16, K = kk+16..31
// C fragment: VGPR r, lane l -> row (r + 8*(l>=16)), col (l&15)
// ---------------------------------------------------------------------------
__global__ void __launch_bounds__(128)
sim_loss_main(const _Float16* __restrict__ g, const int* __restrict__ colLab,
              float* __restrict__ posPart, float* __restrict__ negPart) {
    const int lane  = threadIdx.x & 31;
    const int wave  = threadIdx.x >> 5;        // 0..3
    const int lo16  = lane & 15;
    const int hiSel = lane >> 4;               // 0 or 1
    const int rowBase = blockIdx.x * ROWS_PER_BLOCK + wave * ROWS_PER_WAVE;
    const int chunk   = blockIdx.y;
    const int colChunkBase = chunk * COLS_PER_CHUNK;

    // Row labels for the 32 rows this lane's C registers touch.
    int rlab[4][8];
#pragma unroll
    for (int tm = 0; tm < 4; ++tm)
#pragma unroll
        for (int r = 0; r < 8; ++r)
            rlab[tm][r] = colLab[rowBase + tm * 16 + hiSel * 8 + r];

    float pos[4][8], neg[4][8];
#pragma unroll
    for (int tm = 0; tm < 4; ++tm)
#pragma unroll
        for (int r = 0; r < 8; ++r) { pos[tm][r] = 0.f; neg[tm][r] = 0.f; }

    // Per-lane A row base pointers (include the lane-half K offset).
    const _Float16* aRow[4];
#pragma unroll
    for (int tm = 0; tm < 4; ++tm)
        aRow[tm] = g + (size_t)(rowBase + tm * 16 + lo16) * DIM + hiSel * 16;

    for (int cb = 0; cb < COLS_PER_CHUNK / 64; ++cb) {
        const int colBase = colChunkBase + cb * 64;
        const _Float16* bCol[4];
        int clab[4];
#pragma unroll
        for (int tn = 0; tn < 4; ++tn) {
            int col = colBase + tn * 16 + lo16;
            bCol[tn] = g + (size_t)col * DIM + hiSel * 16;
            clab[tn] = colLab[col];
        }

        v8f acc[4][4];
#pragma unroll
        for (int tm = 0; tm < 4; ++tm)
#pragma unroll
            for (int tn = 0; tn < 4; ++tn)
                acc[tm][tn] = (v8f)0.f;

        for (int ks = 0; ks < DIM / 32; ++ks) {
            const int kk = ks * 32;
            v16h aF[4], bF[4];
#pragma unroll
            for (int tm = 0; tm < 4; ++tm) {
                v8h lo = *(const v8h*)(aRow[tm] + kk);       // K kk+hi*16 .. +7
                v8h hi = *(const v8h*)(aRow[tm] + kk + 8);   // K kk+hi*16+8 .. +15
                aF[tm] = __builtin_shufflevector(lo, hi,
                          0, 1, 2, 3, 4, 5, 6, 7, 8, 9, 10, 11, 12, 13, 14, 15);
            }
#pragma unroll
            for (int tn = 0; tn < 4; ++tn)
                bF[tn] = *(const v16h*)(bCol[tn] + kk);      // 32B contiguous
#pragma unroll
            for (int tm = 0; tm < 4; ++tm)
#pragma unroll
                for (int tn = 0; tn < 4; ++tn)
                    acc[tm][tn] = __builtin_amdgcn_wmma_f32_16x16x32_f16(
                        false, aF[tm], false, bF[tn],
                        (short)0, acc[tm][tn], false, false);
        }

        // Epilogue: one exp per sim element, masked accumulate per row.
#pragma unroll
        for (int tm = 0; tm < 4; ++tm)
#pragma unroll
            for (int tn = 0; tn < 4; ++tn)
#pragma unroll
                for (int r = 0; r < 8; ++r) {
                    float s  = acc[tm][tn][r];
                    bool  isPos = (clab[tn] == rlab[tm][r]);
                    float e  = __expf(isPos ? -s : s);
                    e = (clab[tn] >= 0) ? e : 0.f;           // invalid col: drop
                    pos[tm][r] += isPos ? e : 0.f;
                    neg[tm][r] += isPos ? 0.f : e;
                }
    }

    // Cross-lane reduce within each 16-lane group; one writer per row.
#pragma unroll
    for (int tm = 0; tm < 4; ++tm)
#pragma unroll
        for (int r = 0; r < 8; ++r) {
            float p = pos[tm][r], n = neg[tm][r];
#pragma unroll
            for (int off = 1; off < 16; off <<= 1) {
                p += __shfl_xor(p, off, 32);
                n += __shfl_xor(n, off, 32);
            }
            if (lo16 == r) {
                int row = rowBase + tm * 16 + hiSel * 8 + r;
                posPart[(size_t)row * NCS + chunk] = p;
                negPart[(size_t)row * NCS + chunk] = n;
            }
        }
}

// ---------------------------------------------------------------------------
// Final deterministic reduction: row_loss = log1p(posSum * negSum) for valid
// rows; output = sum / n_valid. Single block, fixed summation order.
// ---------------------------------------------------------------------------
__global__ void finalize(const int* __restrict__ colLab,
                         const float* __restrict__ posPart,
                         const float* __restrict__ negPart,
                         float* __restrict__ out) {
    __shared__ float ssum[256];
    __shared__ int   scnt[256];
    const int tid = threadIdx.x;
    float sum = 0.f;
    int   cnt = 0;
    for (int i = tid; i < N_TOK; i += 256) {
        if (colLab[i] >= 0) {
            float p = 0.f, n = 0.f;
#pragma unroll
            for (int c = 0; c < NCS; ++c) {
                p += posPart[(size_t)i * NCS + c];
                n += negPart[(size_t)i * NCS + c];
            }
            sum += log1pf(p * n);   // == softplus(lse_pos + lse_neg)
            cnt += 1;
        }
    }
    ssum[tid] = sum;
    scnt[tid] = cnt;
    __syncthreads();
    for (int s = 128; s >= 1; s >>= 1) {
        if (tid < s) { ssum[tid] += ssum[tid + s]; scnt[tid] += scnt[tid + s]; }
        __syncthreads();
    }
    if (tid == 0) out[0] = ssum[0] / (float)scnt[0];
}

// ---------------------------------------------------------------------------
extern "C" void kernel_launch(void* const* d_in, const int* in_sizes, int n_in,
                              void* d_out, int out_size, void* d_ws, size_t ws_size,
                              hipStream_t stream) {
    const float* input_f = (const float*)d_in[0];   // [256,27,1024] f32
    const int*   target  = (const int*)d_in[1];     // [256,27] i32
    float* out = (float*)d_out;                     // scalar f32

    char* ws = (char*)d_ws;
    _Float16* g      = (_Float16*)ws;
    int*      colLab = (int*)(ws + WS_OFF_COLLAB);
    float*    posP   = (float*)(ws + WS_OFF_POS);
    float*    negP   = (float*)(ws + WS_OFF_NEG);

    prep_labels<<<1, 256, 0, stream>>>(target, colLab);
    prep_norm<<<N_TOK / 8, 256, 0, stream>>>(input_f, g);

    dim3 grid(N_TOK / ROWS_PER_BLOCK, NCS);         // (27, 27)
    sim_loss_main<<<grid, 128, 0, stream>>>(g, colLab, posP, negP);

    finalize<<<1, 256, 0, stream>>>(colLab, posP, negP, out);
}
